// selective_matching_hor_20280835572218
// MI455X (gfx1250) — matrix operand order repaired and checked by reading, not compiled, not run on hardware
//
#include <hip/hip_runtime.h>
#include <hip/hip_bf16.h>

// ---- problem constants (from reference) ----
#define ANc   5
#define AN2   25
#define Cc    32
#define Kn    6
#define PSH   8
#define Nn    2
#define Hh    128
#define Ww    128
#define PNH   16              // W / PSH
#define Bb    (Nn*PNH)        // 32  distance-batches
#define CM    (Cc*ANc*PSH)    // 1280 rows of M
#define Pp    (ANc*Hh)        // 640  columns of M
#define NQ    (Nn*ANc*Hh)     // 1280 epi "batch" images

typedef __attribute__((ext_vector_type(16))) _Float16 v16h;
typedef __attribute__((ext_vector_type(8)))  float    v8f;

union U16h { v16h v; unsigned int u[8]; };
union Pk2  { _Float16 h[2]; unsigned int u; };

__device__ __forceinline__ float leaky(float x) { return x >= 0.f ? x : 0.1f * x; }

// -------------------------------------------------------------------------
// 1) Pack M into column-major f16  Mb[b][q][cm]  and compute column norms.
//    M[b=(n,pp)][cm=(c,v,s)][q=(u,h)] = lf[(n*25+u*5+v), c, h, pp*8+s]
// -------------------------------------------------------------------------
__global__ void k_pack(const float* __restrict__ lf, _Float16* __restrict__ Mb,
                       float* __restrict__ sq) {
  const int bq = blockIdx.x;            // b*P + q
  const int b  = bq / Pp, q = bq % Pp;
  const int n  = b >> 4, pp = b & 15;
  const int u  = q >> 7, h  = q & 127;
  __shared__ float red[256];
  float acc = 0.f;
  const long colBase = (long)bq * CM;
  for (int cm = threadIdx.x; cm < CM; cm += 256) {
    const int c = cm / 40, r = cm % 40, v = r >> 3, s = r & 7;
    const float val = lf[(((long)(n*AN2 + u*ANc + v)*Cc + c) << 14) + h*Ww + pp*PSH + s];
    Mb[colBase + cm] = (_Float16)val;
    acc += val * val;
  }
  red[threadIdx.x] = acc; __syncthreads();
  for (int st = 128; st > 0; st >>= 1) {
    if ((int)threadIdx.x < st) red[threadIdx.x] += red[threadIdx.x + st];
    __syncthreads();
  }
  if (threadIdx.x == 0) sq[bq] = red[0];
}

// -------------------------------------------------------------------------
// 2) Distance tiles via WMMA:  d[b,p,q] = sq[p] + sq[q] - 2 * (M^T M)[p,q]
//    One wave computes a 16x16 tile, 40 k-steps of v_wmma_f32_16x16x32_f16.
// -------------------------------------------------------------------------
__global__ void k_dist(const _Float16* __restrict__ Mb, const float* __restrict__ sq,
                       float* __restrict__ dist) {
  const int b = blockIdx.z;
  const int pBase = blockIdx.x * 16, qBase = blockIdx.y * 16;
  const int l = threadIdx.x, lm = l & 15, half = l >> 4;
  const unsigned int* MbU = (const unsigned int*)Mb;      // 2 halves / dword
  const long aRow = ((long)b * Pp + (pBase + lm)) * (CM / 2);  // dword units
  const long bRow = ((long)b * Pp + (qBase + lm)) * (CM / 2);
  v8f c = {};
  for (int ks = 0; ks < CM / 32; ++ks) {
    const int kb = ks * 16;                               // 32 halves = 16 dwords
    U16h a, bb;
#pragma unroll
    for (int j = 0; j < 8; ++j) {
      // A operand, contiguous k order used here (same as B): pairs at K = half*16 + 2j
      const int ka = half * 8 + j;    // dword offset
      a.u[j] = MbU[aRow + kb + ka];
      // B operand, contiguous k order used here (same as A): pairs at K = half*16 + 2j
      bb.u[j] = MbU[bRow + kb + half * 8 + j];
    }
    c = __builtin_amdgcn_wmma_f32_16x16x32_f16(false, a.v, false, bb.v,
                                               (short)0, c, false, false);
  }
  const float sqq = sq[b * Pp + qBase + lm];
#pragma unroll
  for (int r = 0; r < 8; ++r) {
    const int p = pBase + r + half * 8;
    dist[((long)b * Pp + p) * Pp + qBase + lm] = sq[b * Pp + p] + sqq - 2.0f * c[r];
  }
}

// -------------------------------------------------------------------------
// 3) Top-6 smallest per row (ascending d, ties keep lower index like top_k).
// -------------------------------------------------------------------------
__global__ void k_topk(const float* __restrict__ dist, int* __restrict__ idx) {
  const int t = blockIdx.x * blockDim.x + threadIdx.x;
  if (t >= Bb * Pp) return;
  const float* row = dist + (long)t * Pp;
  float bd[Kn]; int bi[Kn];
#pragma unroll
  for (int k = 0; k < Kn; ++k) { bd[k] = 3.4e38f; bi[k] = 0; }
  for (int q = 0; q < Pp; ++q) {
    const float d = row[q];
    if (d < bd[Kn - 1]) {
      int k = Kn - 1;
      while (k > 0 && d < bd[k - 1]) { bd[k] = bd[k - 1]; bi[k] = bi[k - 1]; --k; }
      bd[k] = d; bi[k] = q;
    }
  }
#pragma unroll
  for (int k = 0; k < Kn; ++k) idx[t * Kn + k] = bi[k];
}

// -------------------------------------------------------------------------
// 4) Pack w2 (f32, [32][64][3][3]) into the WMMA A-operand lane layout, f16.
//    K index = (dy*3+dx)*64 + ci ; kstep = tap*2 + (ci>=32).
// -------------------------------------------------------------------------
__global__ void k_w2pack(const float* __restrict__ w2, unsigned int* __restrict__ Ap) {
  const int t = blockIdx.x * blockDim.x + threadIdx.x;
  if (t >= 2 * 18 * 32 * 8) return;
  const int j = t & 7; int r = t >> 3;
  const int l = r & 31; r >>= 5;
  const int ks = r % 18, mt = r / 18;
  const int half = l >> 4, lm = l & 15;
  const int co = mt * 16 + lm;
  const int koff = half * 8 + ((j < 4) ? j * 2 : 16 + (j - 4) * 2);
  const int tap = ks >> 1, ci = (ks & 1) * 32 + koff;
  const int dy = tap / 3, dx = tap % 3;
  Pk2 pk;
  pk.h[0] = (_Float16)w2[((co * 64 + ci) * 3 + dy) * 3 + dx];
  pk.h[1] = (_Float16)w2[((co * 64 + ci + 1) * 3 + dy) * 3 + dx];
  Ap[t] = pk.u;
}

// -------------------------------------------------------------------------
// 5) Gather + 1x1 conv (w1: 32 x 192) + leaky.  One block per (n,u,h,v,pp);
//    gathered 192x8 patch staged in LDS, output sel1[(n,u,h)][co][v][w].
// -------------------------------------------------------------------------
__global__ void k_conv1(const float* __restrict__ lf, const float* __restrict__ w1,
                        const int* __restrict__ idx, float* __restrict__ sel1) {
  int bid = blockIdx.x;
  const int pp = bid & 15; bid >>= 4;
  const int v = bid % 5;   bid /= 5;
  const int h = bid & 127; bid >>= 7;
  const int u = bid % 5;
  const int n = bid / 5;
  __shared__ float w1s[Cc * Cc * Kn];   // 32 * 192 = 6144 floats (24 KB)
  __shared__ float g[192 * 8];          // gathered patch (6 KB)
  __shared__ int   js[Kn];
  for (int i = threadIdx.x; i < Cc * Cc * Kn; i += 256) w1s[i] = w1[i];
  if ((int)threadIdx.x < Kn)
    js[threadIdx.x] = idx[((long)((n * PNH + pp) * Pp + u * Hh + h)) * Kn + threadIdx.x];
  __syncthreads();
  for (int e = threadIdx.x; e < 192 * 8; e += 256) {
    const int s = e & 7, kcc = e >> 3, k = kcc >> 5, cc = kcc & 31;
    const int j = js[k], uj = j >> 7, hj = j & 127;
    g[e] = lf[(((long)(n * AN2 + uj * ANc + v) * Cc + cc) << 14) + hj * Ww + pp * PSH + s];
  }
  __syncthreads();
  const int co = threadIdx.x >> 3, s = threadIdx.x & 7;
  float acc = 0.f;
#pragma unroll 8
  for (int kcc = 0; kcc < 192; ++kcc) acc += w1s[co * 192 + kcc] * g[kcc * 8 + s];
  acc = leaky(acc);
  const int nq = (n * ANc + u) * Hh + h;
  sel1[(((long)nq * Cc + co) * ANc + v) * Ww + pp * PSH + s] = acc;
}

// -------------------------------------------------------------------------
// 6) 3x3 conv (64 -> 32) as im2col GEMM via WMMA, + leaky + output permute.
//    One wave per 16x16 output tile; 18 k-steps; B operand built on the fly
//    from lf_fea (ci<32, lf_ver view) or sel1 (ci>=32) with zero padding.
//    Output written directly as out[(n*25+u*5+v)][co][h][w].
// -------------------------------------------------------------------------
__global__ void k_conv3(const float* __restrict__ lf, const float* __restrict__ sel1,
                        const unsigned int* __restrict__ Ap, float* __restrict__ out) {
  const int mt = blockIdx.y;
  const int nq = blockIdx.z;
  const int colBase = blockIdx.x * 16;
  const int l = threadIdx.x, lm = l & 15, half = l >> 4;
  const int n = nq / 640, ru = nq % 640, u = ru >> 7, h = ru & 127;
  const int col = colBase + lm, v = col >> 7, w = col & 127;
  v8f c = {};
  for (int ks = 0; ks < 18; ++ks) {
    const int tap = ks >> 1, dy = tap / 3, dx = tap % 3;
    const int y = v + dy - 1, x = w + dx - 1;
    const bool useSel = (ks & 1);
    U16h a, bb;
#pragma unroll
    for (int j = 0; j < 8; ++j) a.u[j] = Ap[((mt * 18 + ks) * 32 + l) * 8 + j];
    const bool valid = (y >= 0) & (y < ANc) & (x >= 0) & (x < Ww);
#pragma unroll
    for (int j = 0; j < 8; ++j) {
      unsigned int d = 0u;
      if (valid) {
        const int cc = half * 8 + ((j < 4) ? j * 2 : 16 + (j - 4) * 2);          // channel within 32-block
        float f0, f1;
        if (!useSel) {
          const long base = (((long)(n * AN2 + u * ANc + y) * Cc + cc) << 14) + h * Ww + x;
          f0 = lf[base]; f1 = lf[base + (1 << 14)];
        } else {
          const long base = (((long)nq * Cc + cc) * ANc + y) * Ww + x;
          f0 = sel1[base]; f1 = sel1[base + ANc * Ww];
        }
        Pk2 pk; pk.h[0] = (_Float16)f0; pk.h[1] = (_Float16)f1; d = pk.u;
      }
      bb.u[j] = d;
    }
    c = __builtin_amdgcn_wmma_f32_16x16x32_f16(false, a.v, false, bb.v,
                                               (short)0, c, false, false);
  }
#pragma unroll
  for (int r = 0; r < 8; ++r) {
    const float o = leaky(c[r]);
    const int co = mt * 16 + r + half * 8;
    out[(((long)(n * AN2 + u * ANc + v) * Cc + co) << 14) + h * Ww + w] = o;
  }
}

// -------------------------------------------------------------------------
// launcher — workspace layout (bytes, all 256B aligned):
//   Mb   f16  [32][640][1280]   52,428,800
//   sq   f32  [32*640]              81,920
//   dist f32  [32][640][640]    52,428,800
//   idx  i32  [32*640][6]          491,520
//   sel1 f32  [1280][32][5][128] 104,857,600
//   Ap   u32  [2][18][32][8]        36,864
// -------------------------------------------------------------------------
extern "C" void kernel_launch(void* const* d_in, const int* in_sizes, int n_in,
                              void* d_out, int out_size, void* d_ws, size_t ws_size,
                              hipStream_t stream) {
  const float* lf = (const float*)d_in[0];
  const float* w1 = (const float*)d_in[1];
  const float* w2 = (const float*)d_in[2];
  float* out = (float*)d_out;
  char* ws = (char*)d_ws;

  _Float16*    Mb   = (_Float16*)(ws);
  float*       sq   = (float*)(ws + 52428800);
  float*       dist = (float*)(ws + 52510720);
  int*         idx  = (int*)(ws + 104939520);
  float*       sel1 = (float*)(ws + 105431040);
  unsigned int* Ap  = (unsigned int*)(ws + 115916800);

  k_pack<<<Bb * Pp, 256, 0, stream>>>(lf, Mb, sq);
  k_dist<<<dim3(Pp / 16, Pp / 16, Bb), 32, 0, stream>>>(Mb, sq, dist);
  k_topk<<<(Bb * Pp + 255) / 256, 256, 0, stream>>>(dist, idx);
  k_w2pack<<<(2 * 18 * 32 * 8 + 255) / 256, 256, 0, stream>>>(w2, Ap);
  k_conv1<<<Nn * ANc * Hh * ANc * PNH, 256, 0, stream>>>(lf, w1, idx, sel1);
  k_conv3<<<dim3(Pp / 16, 2, NQ), 32, 0, stream>>>(lf, sel1, Ap, out);
}
